// HyperbolicRotHRel_69303592288543
// MI455X (gfx1250) — matrix operand order repaired and checked by reading, not compiled, not run on hardware
//
#include <hip/hip_runtime.h>
#include <hip/hip_bf16.h>

typedef __attribute__((ext_vector_type(16))) __bf16 v16bf;
typedef __attribute__((ext_vector_type(8)))  __bf16 v8bf;
typedef __attribute__((ext_vector_type(8)))  float  v8f;
typedef __attribute__((ext_vector_type(4)))  unsigned int v4u;
typedef __attribute__((ext_vector_type(8)))  int v8i;
typedef __attribute__((ext_vector_type(4)))  int v4i;

#define D_DIM 512          // embedding dim (compile-time: fragment layouts depend on it)
#define HCC   0.01f        // curvature c
#define MINN  1e-15f

// ---------------------------------------------------------------------------
// Kernel 1: rel_hyp = exp_map_zero(rel_embedding, c); store bf16 hi/lo + y2
// ---------------------------------------------------------------------------
__global__ void __launch_bounds__(256)
rel_transform_kernel(const float* __restrict__ rel,
                     __bf16* __restrict__ Yhi, __bf16* __restrict__ Ylo,
                     float* __restrict__ y2out) {
  const int r = blockIdx.x;
  const int t = threadIdx.x;                 // 0..255
  const float* row = rel + (size_t)r * D_DIM;
  const float a = row[t];
  const float b = row[t + 256];

  __shared__ float sred[256];
  sred[t] = a * a + b * b;
  __syncthreads();
  for (int s = 128; s > 0; s >>= 1) {
    if (t < s) sred[t] += sred[t + s];
    __syncthreads();
  }
  const float n2 = sred[0];
  const float n  = fmaxf(sqrtf(n2), MINN);
  const float sc = tanhf(0.1f * n) / (0.1f * n);   // sqrt(c)=0.1
  if (t == 0) y2out[r] = sc * sc * n2;

  const float ya = sc * a, yb = sc * b;
  const __bf16 ha = (__bf16)ya, hb = (__bf16)yb;
  const size_t base = (size_t)r * D_DIM;
  Yhi[base + t]       = ha;
  Yhi[base + t + 256] = hb;
  Ylo[base + t]       = (__bf16)(ya - (float)ha);
  Ylo[base + t + 256] = (__bf16)(yb - (float)hb);
}

// ---------------------------------------------------------------------------
// Kernel 2: gather s/o, log-map, Givens rotation, exp-map, first mobius_add.
// Produces X = -query (bf16 hi/lo) and x2[b] = ||query||^2 (analytic).
// ---------------------------------------------------------------------------
__global__ void __launch_bounds__(256)
query_kernel(const float* __restrict__ ent, const int* __restrict__ trip,
             const float* __restrict__ grot,
             __bf16* __restrict__ Xhi, __bf16* __restrict__ Xlo,
             float* __restrict__ x2out) {
  const int b = blockIdx.x;
  const int t = threadIdx.x;                 // pair index 0..255
  const int si = trip[3 * b + 0];
  const int oi = trip[3 * b + 2];
  const float2 s = ((const float2*)(ent + (size_t)si * D_DIM))[t];
  const float2 o = ((const float2*)(ent + (size_t)oi * D_DIM))[t];

  __shared__ float r0[256];
  __shared__ float r1[256];
  r0[t] = s.x * s.x + s.y * s.y;
  r1[t] = o.x * o.x + o.y * o.y;
  __syncthreads();
  for (int k = 128; k > 0; k >>= 1) {
    if (t < k) { r0[t] += r0[t + k]; r1[t] += r1[t + k]; }
    __syncthreads();
  }
  const float ss = r0[0];
  const float oo = r1[0];
  __syncthreads();

  // log_map_zero(s)
  const float ns  = fmaxf(sqrtf(ss), MINN);
  const float arg = fminf(0.1f * ns, 1.0f - 1e-7f);
  const float sc1 = atanhf(arg) / (0.1f * ns);
  const float x1 = sc1 * s.x, x2e = sc1 * s.y;

  // Givens rotation
  const float ang = grot[t];
  const float ca = cosf(ang), sa = sinf(ang);
  const float v0 = ca * x1 - sa * x2e;
  const float v1 = sa * x1 + ca * x2e;

  r0[t] = v0 * v0 + v1 * v1;     // ||rot||^2
  r1[t] = v0 * o.x + v1 * o.y;   // <rot, o>
  __syncthreads();
  for (int k = 128; k > 0; k >>= 1) {
    if (t < k) { r0[t] += r0[t + k]; r1[t] += r1[t + k]; }
    __syncthreads();
  }
  const float vv = r0[0];
  const float vo = r1[0];

  // exp_map_zero(rot)
  const float nv  = fmaxf(sqrtf(vv), MINN);
  const float sc2 = tanhf(0.1f * nv) / (0.1f * nv);

  // mobius_add(x = -rot_s, y = o)
  const float xx = sc2 * sc2 * vv;
  const float xy = -sc2 * vo;
  const float yy = oo;
  const float A   = 1.0f + 2.0f * HCC * xy + HCC * yy;
  const float Bf  = 1.0f - HCC * xx;
  const float den = fmaxf(1.0f + 2.0f * HCC * xy + HCC * HCC * xx * yy, MINN);
  const float inv = 1.0f / den;

  const float q0 = (A * (-sc2 * v0) + Bf * o.x) * inv;
  const float q1 = (A * (-sc2 * v1) + Bf * o.y) * inv;
  if (t == 0)
    x2out[b] = (A * A * xx + 2.0f * A * Bf * xy + Bf * Bf * yy) * inv * inv;

  const float X0 = -q0, X1 = -q1;
  const __bf16 h0 = (__bf16)X0, h1 = (__bf16)X1;
  const size_t base = (size_t)b * D_DIM + 2 * t;
  Xhi[base]     = h0;
  Xhi[base + 1] = h1;
  Xlo[base]     = (__bf16)(X0 - (float)h0);
  Xlo[base + 1] = (__bf16)(X1 - (float)h1);
}

// ---------------------------------------------------------------------------
// TDM helper: 2-D tile load Global -> LDS via Tensor Data Mover.
// Builds D# groups per CDNA5 ISA and issues
// tensor_load_to_lds (6-arg builtin on this toolchain). rows x width bf16.
// ---------------------------------------------------------------------------
__device__ __forceinline__ void tdm_load_tile_bf16(unsigned lds_off,
                                                   const __bf16* gsrc,
                                                   unsigned width_elems,
                                                   unsigned rows,
                                                   unsigned stride_elems) {
  const unsigned long long ga = (unsigned long long)(size_t)gsrc;
  v4u g0;
  g0[0] = 1u;                                            // count=1, user desc
  g0[1] = lds_off;                                       // lds_addr
  g0[2] = (unsigned)(ga & 0xFFFFFFFFull);                // global_addr[31:0]
  g0[3] = (unsigned)((ga >> 32) & 0x01FFFFFFull)         // global_addr[56:32]
          | (2u << 30);                                  // type = 2 (image)
  v8i g1;
  g1[0] = (int)(1u << 16);                               // data_size=1 (2B), wg_mask=0
  g1[1] = (int)((width_elems & 0xFFFFu) << 16);          // tensor_dim0[15:0]
  g1[2] = (int)(((width_elems >> 16) & 0xFFFFu)          // tensor_dim0[31:16]
          | ((rows & 0xFFFFu) << 16));                   // tensor_dim1[15:0]
  g1[3] = (int)(((rows >> 16) & 0xFFFFu)                 // tensor_dim1[31:16]
          | ((width_elems & 0xFFFFu) << 16));            // tile_dim0
  g1[4] = (int)(rows & 0xFFFFu);                         // tile_dim1 (tile_dim2=0)
  g1[5] = (int)stride_elems;                             // tensor_dim0_stride[31:0]
  g1[6] = 0;                                             // stride[47:32], dim1_stride
  g1[7] = 0;
  v4i gz4 = {0, 0, 0, 0};
  v8i gz8 = {0, 0, 0, 0, 0, 0, 0, 0};
  __builtin_amdgcn_tensor_load_to_lds(g0, g1, gz4, gz4, gz8, 0);
}

// ---------------------------------------------------------------------------
// Kernel 3: xy = X @ Y^T via split-bf16 WMMA (hi*hi + hi*lo + lo*hi), fused
// Mobius-distance epilogue. Block = 4 waves covering 16(M) x 128(N); the
// block's shared 16x512 A panel (hi+lo) is staged in LDS once via the TDM.
// ---------------------------------------------------------------------------
__global__ void __launch_bounds__(128)
score_gemm_kernel(const __bf16* __restrict__ Xhi, const __bf16* __restrict__ Xlo,
                  const __bf16* __restrict__ Yhi, const __bf16* __restrict__ Ylo,
                  const float* __restrict__ x2, const float* __restrict__ y2,
                  const float* __restrict__ bias, float* __restrict__ out,
                  int NR) {
  __shared__ __bf16 shAhi[16 * D_DIM];   // 16 KB
  __shared__ __bf16 shAlo[16 * D_DIM];   // 16 KB

  const int wave = threadIdx.x >> 5;
  const int lane = threadIdx.x & 31;
  const int lh = lane & 15;        // row/col within half-wave
  const int lg = lane >> 4;        // which half-wave
  const int mt = blockIdx.x * 16;
  const int nt = blockIdx.y * 128 + wave * 32;

  // Stage the block's A panel (rows mt..mt+15, hi & lo) into LDS via TDM.
  if (wave == 0) {
    const unsigned offHi = (unsigned)(size_t)&shAhi[0];
    const unsigned offLo = (unsigned)(size_t)&shAlo[0];
    tdm_load_tile_bf16(offHi, Xhi + (size_t)mt * D_DIM, D_DIM, 16, D_DIM);
    tdm_load_tile_bf16(offLo, Xlo + (size_t)mt * D_DIM, D_DIM, 16, D_DIM);
    __builtin_amdgcn_s_wait_tensorcnt(0);
  }
  __syncthreads();

  // A fragment (16x32 bf16) from LDS: lane holds M = lh; K pattern:
  //   elems j -> k0 + 16*lg + j
  const __bf16* lah = &shAhi[(size_t)lh * D_DIM];
  const __bf16* lal = &shAlo[(size_t)lh * D_DIM];
  // B fragment (32x16 bf16): lane holds N = nt+lh; elems j -> k0 + 16*lg + j
  const __bf16* b0h = Yhi + (size_t)(nt + lh) * D_DIM;
  const __bf16* b0l = Ylo + (size_t)(nt + lh) * D_DIM;
  const __bf16* b1h = Yhi + (size_t)(nt + 16 + lh) * D_DIM;
  const __bf16* b1l = Ylo + (size_t)(nt + 16 + lh) * D_DIM;

  v8f acc0 = {0.f, 0.f, 0.f, 0.f, 0.f, 0.f, 0.f, 0.f};
  v8f acc1 = acc0;

  const int kA = 16 * lg;
  const int kB = 16 * lg;

  for (int k0 = 0; k0 < D_DIM; k0 += 32) {
    __builtin_prefetch(b0h + k0 + 64, 0, 1);
    __builtin_prefetch(b1h + k0 + 64, 0, 1);

    v8bf aL  = *(const v8bf*)(lah + k0 + kA);
    v8bf aH  = *(const v8bf*)(lah + k0 + kA + 8);
    v16bf Ah = __builtin_shufflevector(aL, aH, 0,1,2,3,4,5,6,7,8,9,10,11,12,13,14,15);
    v8bf lL  = *(const v8bf*)(lal + k0 + kA);
    v8bf lH  = *(const v8bf*)(lal + k0 + kA + 8);
    v16bf Al = __builtin_shufflevector(lL, lH, 0,1,2,3,4,5,6,7,8,9,10,11,12,13,14,15);

    v16bf B0h = *(const v16bf*)(b0h + k0 + kB);
    v16bf B0l = *(const v16bf*)(b0l + k0 + kB);
    v16bf B1h = *(const v16bf*)(b1h + k0 + kB);
    v16bf B1l = *(const v16bf*)(b1l + k0 + kB);

    acc0 = __builtin_amdgcn_wmma_f32_16x16x32_bf16(false, Ah, false, B0h, (short)0, acc0, false, false);
    acc0 = __builtin_amdgcn_wmma_f32_16x16x32_bf16(false, Ah, false, B0l, (short)0, acc0, false, false);
    acc0 = __builtin_amdgcn_wmma_f32_16x16x32_bf16(false, Al, false, B0h, (short)0, acc0, false, false);

    acc1 = __builtin_amdgcn_wmma_f32_16x16x32_bf16(false, Ah, false, B1h, (short)0, acc1, false, false);
    acc1 = __builtin_amdgcn_wmma_f32_16x16x32_bf16(false, Ah, false, B1l, (short)0, acc1, false, false);
    acc1 = __builtin_amdgcn_wmma_f32_16x16x32_bf16(false, Al, false, B1h, (short)0, acc1, false, false);
  }

  // Epilogue: C/D layout — VGPR g, lane L: M = g + 8*(L>=16), N = L&15
  #pragma unroll
  for (int tile = 0; tile < 2; ++tile) {
    const int r = nt + tile * 16 + lh;
    const float yy = y2[r];
    const float bb = bias[r];
    const v8f acc = tile ? acc1 : acc0;
    #pragma unroll
    for (int g = 0; g < 8; ++g) {
      const int row = mt + g + 8 * lg;
      const float xy = acc[g];
      const float xx = x2[row];
      const float A   = 1.0f + 2.0f * HCC * xy + HCC * yy;
      const float Bf  = 1.0f - HCC * xx;
      const float den = fmaxf(1.0f + 2.0f * HCC * xy + HCC * HCC * xx * yy, MINN);
      const float inv = 1.0f / den;
      const float dist = (A * A * xx + 2.0f * A * Bf * xy + Bf * Bf * yy) * inv * inv;
      out[(size_t)row * NR + r] = bb - dist;
    }
  }
}

// ---------------------------------------------------------------------------
extern "C" void kernel_launch(void* const* d_in, const int* in_sizes, int n_in,
                              void* d_out, int out_size, void* d_ws, size_t ws_size,
                              hipStream_t stream) {
  const float* ent  = (const float*)d_in[0];   // (20000, 512) f32
  const float* rel  = (const float*)d_in[1];   // (512, 512)   f32
  const int*   trip = (const int*)d_in[2];     // (1024, 3)    int
  const float* grot = (const float*)d_in[3];   // (256,)       f32
  const float* bias = (const float*)d_in[4];   // (512,)       f32
  float* out = (float*)d_out;                  // (1024, 512)  f32

  const int B = in_sizes[2] / 3;               // 1024
  const int R = in_sizes[4];                   // 512

  char* ws = (char*)d_ws;
  size_t off = 0;
  auto alloc = [&](size_t bytes) -> void* {
    void* p = ws + off;
    off = (off + bytes + 255) & ~(size_t)255;
    return p;
  };
  __bf16* Xhi = (__bf16*)alloc((size_t)B * D_DIM * sizeof(__bf16));
  __bf16* Xlo = (__bf16*)alloc((size_t)B * D_DIM * sizeof(__bf16));
  __bf16* Yhi = (__bf16*)alloc((size_t)R * D_DIM * sizeof(__bf16));
  __bf16* Ylo = (__bf16*)alloc((size_t)R * D_DIM * sizeof(__bf16));
  float*  x2  = (float*)alloc((size_t)B * sizeof(float));
  float*  y2  = (float*)alloc((size_t)R * sizeof(float));

  rel_transform_kernel<<<R, 256, 0, stream>>>(rel, Yhi, Ylo, y2);
  query_kernel<<<B, 256, 0, stream>>>(ent, trip, grot, Xhi, Xlo, x2);

  dim3 grid(B / 16, R / 128);
  score_gemm_kernel<<<grid, 128, 0, stream>>>(Xhi, Xlo, Yhi, Ylo, x2, y2, bias,
                                              out, R);
}
